// Gpt2MambaSelectiveBlock_71829033059047
// MI455X (gfx1250) — matrix operand order repaired and checked by reading, not compiled, not run on hardware
//
#include <hip/hip_runtime.h>
#include <hip/hip_bf16.h>
#include <math.h>

// ---------------- problem sizes ----------------
#define HIDN   768
#define NHEAD  12
#define HDIM   64
#define INTC   1536
#define NSTATE 16
#define RDTC   48
#define XDC    80            // RDT + 2*NST
#define BBATCH 2
#define TSEQ   1024
#define MROWS  (BBATCH*TSEQ) // 2048
#define LNEPS  1e-5f
#define GEMM_WAVES 4

typedef __bf16 bf16_t;
typedef __attribute__((ext_vector_type(16))) __bf16 v16bf;
typedef __attribute__((ext_vector_type(8)))  __bf16 v8bf;
typedef __attribute__((ext_vector_type(8)))  float  v8f;
typedef __attribute__((ext_vector_type(4)))  unsigned int v4u;
typedef __attribute__((ext_vector_type(8)))  int v8i;
typedef __attribute__((ext_vector_type(4)))  int v4i;

__device__ __forceinline__ bf16_t f2bf(float f) { return (bf16_t)f; }

__device__ __forceinline__ v16bf pack16(v8bf lo, v8bf hi) {
  v16bf a;
#pragma unroll
  for (int i = 0; i < 8; ++i) { a[i] = lo[i]; a[i + 8] = hi[i]; }
  return a;
}

__device__ __forceinline__ v8f wmma_bf16(v16bf a, v16bf b, v8f c) {
  // (neg_a, A, neg_b, B, c_mod, C, reuse_a, reuse_b)
  return __builtin_amdgcn_wmma_f32_16x16x32_bf16(false, a, false, b, (short)0, c,
                                                 false, false);
}

// TDM: DMA a [rows=16, K] bf16 row-major panel at `gaddr` into LDS offset 0.
// D# layout per CDNA5 ISA (2-D tensor, groups 2/3 zero).
__device__ __forceinline__ void tdm_load_panel16(const bf16_t* gaddr, int K) {
  unsigned long long ga = (unsigned long long)gaddr;
  v4u g0;
  g0[0] = 1u;                                   // count=1, user mode
  g0[1] = 0u;                                   // lds_addr = 0 (dynamic LDS base)
  g0[2] = (unsigned int)(ga & 0xffffffffu);     // global_addr[31:0]
  g0[3] = (unsigned int)((ga >> 32) & 0x01ffffffu) | 0x80000000u; // addr[56:32] | type=2
  v8i g1;
  g1[0] = (int)(1u << 16);                      // wg_mask=0, data_size=1 (2 bytes)
  g1[1] = (int)((unsigned)(K & 0xffff) << 16);  // tensor_dim0[15:0] @ bits 63:48
  g1[2] = (int)(((unsigned)K >> 16) | (16u << 16)); // tensor_dim0 hi | tensor_dim1=16
  g1[3] = (int)((unsigned)(K & 0xffff) << 16);  // tile_dim0 = K @ bits 127:112
  g1[4] = 16;                                   // tile_dim1 = 16, tile_dim2 = 0
  g1[5] = K;                                    // tensor_dim0_stride[31:0]
  g1[6] = 0;                                    // stride hi, tensor_dim1_stride lo
  g1[7] = 0;
  v4i gz = {0, 0, 0, 0};
#if __clang_major__ >= 23
  v8i gz8 = {0, 0, 0, 0, 0, 0, 0, 0};
  __builtin_amdgcn_tensor_load_to_lds(g0, g1, gz, gz, gz8, 0);
#else
  __builtin_amdgcn_tensor_load_to_lds(g0, g1, gz, gz, 0);
#endif
  __builtin_amdgcn_s_wait_tensorcnt((short)0);
}

// ---------------- weight transpose + cast: W[K,N] f32 -> Wt[N,Kpad] bf16 ----
__global__ void k_wcast(const float* __restrict__ W, bf16_t* __restrict__ Wt,
                        int K, int N, int Kpad) {
  long long idx = (long long)blockIdx.x * blockDim.x + threadIdx.x;
  long long tot = (long long)N * Kpad;
  if (idx >= tot) return;
  int n  = (int)(idx / Kpad);
  int kp = (int)(idx % Kpad);
  Wt[idx] = (kp < K) ? f2bf(W[(size_t)kp * N + n]) : f2bf(0.f);
}

// ---------------- generic WMMA GEMM ----------------------------------------
// A[M,K] bf16 row-major (global, streamed), Bt[N,K] bf16 row-major weight.
// The block's 16-row B panel is DMA'd to LDS once by the Tensor Data Mover and
// shared by all GEMM_WAVES waves; A fragments stream from global with prefetch.
// ACT: 0 none, 1 softplus, 2 exact gelu.  OUTBF: write bf16 instead of f32.
template <int ACT, int HAS_BIAS, int HAS_RES, int OUTBF>
__global__ void k_gemm(const bf16_t* __restrict__ A, const bf16_t* __restrict__ Bt,
                       void* __restrict__ Cv, const float* __restrict__ bias,
                       const float* __restrict__ resid, int N, int K) {
  extern __shared__ char smemB[];
  bf16_t* sB = (bf16_t*)smemB;                  // [16][K] bf16 weight panel
  const int lane = threadIdx.x & 31;
  const int wave = threadIdx.x >> 5;
  const int n0 = blockIdx.x * 16;
  const int m0 = (blockIdx.y * GEMM_WAVES + wave) * 16;
  const int half = lane >> 4, lh = lane & 15;
  // stage B panel [n0..n0+15, 0..K) into LDS via TDM (one op, wave 0 only)
  if (wave == 0) tdm_load_panel16(Bt + (size_t)n0 * K, K);
  __syncthreads();
  const bf16_t* arow = A + (size_t)(m0 + lh) * K;
  const bf16_t* brow = sB + (size_t)lh * K;
  v8f acc = {};
  for (int k0 = 0; k0 < K; k0 += 32) {
    v8bf alo = *(const v8bf*)(arow + k0 + half * 16);
    v8bf ahi = *(const v8bf*)(arow + k0 + half * 16 + 8);
    v16bf b  = *(const v16bf*)(brow + k0 + half * 16);   // ds_load_b128 x2
    if (k0 + 32 < K)
      __builtin_prefetch(arow + k0 + 64, 0, 0);          // global_prefetch_b8
    acc = wmma_bf16(pack16(alo, ahi), b, acc);
  }
#pragma unroll
  for (int r = 0; r < 8; ++r) {
    const int row = m0 + half * 8 + r;
    const int col = n0 + lh;
    float v = acc[r];
    if (HAS_BIAS) v += bias[col];
    if (ACT == 1) v = (v > 20.f) ? v : log1pf(expf(v));                 // softplus
    if (ACT == 2) v = 0.5f * v * (1.f + erff(v * 0.70710678118654752f)); // gelu
    if (HAS_RES) v += resid[(size_t)row * N + col];
    if (OUTBF) ((bf16_t*)Cv)[(size_t)row * N + col] = f2bf(v);
    else       ((float*)Cv)[(size_t)row * N + col] = v;
  }
}

// ---------------- fused LN1 / mamba-LN / rms-norm (one row per block) -------
__global__ void k_ln1(const float* __restrict__ x,
                      const float* __restrict__ g1, const float* __restrict__ b1,
                      const float* __restrict__ gm, const float* __restrict__ bm,
                      const float* __restrict__ rw,
                      bf16_t* __restrict__ xn, float* __restrict__ mx,
                      bf16_t* __restrict__ hb) {
  const int m = blockIdx.x;
  const int t = threadIdx.x;
  const float* xr = x + (size_t)m * HIDN;
  __shared__ float red[256];
  float s = 0.f;
  for (int c = t; c < HIDN; c += 256) s += xr[c];
  red[t] = s; __syncthreads();
  for (int o = 128; o > 0; o >>= 1) { if (t < o) red[t] += red[t + o]; __syncthreads(); }
  const float mean = red[0] / HIDN; __syncthreads();
  float v = 0.f;
  for (int c = t; c < HIDN; c += 256) { float d = xr[c] - mean; v += d * d; }
  red[t] = v; __syncthreads();
  for (int o = 128; o > 0; o >>= 1) { if (t < o) red[t] += red[t + o]; __syncthreads(); }
  const float rinv = rsqrtf(red[0] / HIDN + LNEPS); __syncthreads();
  float s2 = 0.f;
  for (int c = t; c < HIDN; c += 256) {
    float xc = (xr[c] - mean) * rinv;
    xn[(size_t)m * HIDN + c] = f2bf(xc * g1[c] + b1[c]);
    float mv = xc * gm[c] + bm[c];
    mx[(size_t)m * HIDN + c] = mv;
    s2 += mv * mv;
  }
  red[t] = s2; __syncthreads();
  for (int o = 128; o > 0; o >>= 1) { if (t < o) red[t] += red[t + o]; __syncthreads(); }
  const float r2 = rsqrtf(red[0] / HIDN + LNEPS); __syncthreads();
  for (int c = t; c < HIDN; c += 256)
    hb[(size_t)m * HIDN + c] = f2bf(mx[(size_t)m * HIDN + c] * r2 * rw[c]);
}

// ---------------- plain layernorm -> bf16 -----------------------------------
__global__ void k_ln2(const float* __restrict__ x, const float* __restrict__ g,
                      const float* __restrict__ b, bf16_t* __restrict__ o) {
  const int m = blockIdx.x;
  const int t = threadIdx.x;
  const float* xr = x + (size_t)m * HIDN;
  __shared__ float red[256];
  float s = 0.f;
  for (int c = t; c < HIDN; c += 256) s += xr[c];
  red[t] = s; __syncthreads();
  for (int oo = 128; oo > 0; oo >>= 1) { if (t < oo) red[t] += red[t + oo]; __syncthreads(); }
  const float mean = red[0] / HIDN; __syncthreads();
  float v = 0.f;
  for (int c = t; c < HIDN; c += 256) { float d = xr[c] - mean; v += d * d; }
  red[t] = v; __syncthreads();
  for (int oo = 128; oo > 0; oo >>= 1) { if (t < oo) red[t] += red[t + oo]; __syncthreads(); }
  const float rinv = rsqrtf(red[0] / HIDN + LNEPS);
  for (int c = t; c < HIDN; c += 256)
    o[(size_t)m * HIDN + c] = f2bf((xr[c] - mean) * rinv * g[c] + b[c]);
}

// ---------------- causal depthwise conv1d (k=4) + silu ----------------------
__global__ void k_conv(const float* __restrict__ proj, const float* __restrict__ cw,
                       const float* __restrict__ cb, float* __restrict__ hs,
                       bf16_t* __restrict__ hsb) {
  const int i = blockIdx.x * 256 + threadIdx.x;
  const int m = blockIdx.y;
  if (i >= INTC) return;
  const int b = m / TSEQ, t = m % TSEQ;
  float acc = cb[i];
#pragma unroll
  for (int k = 0; k < 4; ++k) {
    int tt = t - 3 + k;
    if (tt >= 0) acc += cw[i * 4 + k] * proj[((size_t)(b * TSEQ + tt)) * (2 * INTC) + i];
  }
  float sv = acc / (1.f + expf(-acc));      // silu
  hs[(size_t)m * INTC + i]  = sv;
  hsb[(size_t)m * INTC + i] = f2bf(sv);
}

// ---------------- dt slice -> bf16, K padded 48 -> 64 ------------------------
__global__ void k_dtprep(const float* __restrict__ ssm, bf16_t* __restrict__ dtb) {
  const int idx = blockIdx.x * 256 + threadIdx.x;
  if (idx >= MROWS * 64) return;
  const int m = idx >> 6, c = idx & 63;
  dtb[idx] = (c < RDTC) ? f2bf(ssm[(size_t)m * XDC + c]) : f2bf(0.f);
}

// ---------------- selective-scan recurrence + gating -------------------------
__global__ void k_scan(const float* __restrict__ dtf, const float* __restrict__ ssm,
                       const float* __restrict__ hs, const float* __restrict__ proj,
                       const float* __restrict__ A_log, const float* __restrict__ Dp,
                       bf16_t* __restrict__ yb) {
  const int gid = blockIdx.x * 256 + threadIdx.x;
  if (gid >= BBATCH * INTC) return;
  const int b = gid / INTC, i = gid % INTC;
  float A[NSTATE], h[NSTATE];
#pragma unroll
  for (int n = 0; n < NSTATE; ++n) { A[n] = -expf(A_log[i * NSTATE + n]); h[n] = 0.f; }
  const float D = Dp[i];
  for (int t = 0; t < TSEQ; ++t) {
    const size_t m = (size_t)b * TSEQ + t;
    const float dt = dtf[m * INTC + i];
    const float u  = hs[m * INTC + i];
    const float g  = proj[m * (2 * INTC) + INTC + i];
    const float* Bm = ssm + m * XDC + RDTC;
    const float* Cm = Bm + NSTATE;
    float y = u * D;
#pragma unroll
    for (int n = 0; n < NSTATE; ++n) {
      h[n] = __expf(dt * A[n]) * h[n] + dt * Bm[n] * u;
      y += h[n] * Cm[n];
    }
    y *= g / (1.f + expf(-g));              // * silu(gate)
    yb[m * INTC + i] = f2bf(y);
  }
}

// ---------------- selectivity scalar per token -------------------------------
__global__ void k_sel(const float* __restrict__ mo, const float* __restrict__ sw,
                      const float* __restrict__ sb, float* __restrict__ sel) {
  const int m = blockIdx.x;
  const int t = threadIdx.x;
  __shared__ float red[256];
  float s = 0.f;
  for (int c = t; c < HIDN; c += 256) {
    float v = mo[(size_t)m * HIDN + c];
    v = fminf(20.f, fmaxf(-20.f, v));
    s += v * sw[c];
  }
  red[t] = s; __syncthreads();
  for (int o = 128; o > 0; o >>= 1) { if (t < o) red[t] += red[t + o]; __syncthreads(); }
  if (t == 0) sel[m] = 1.f / (1.f + expf(-(red[0] + sb[0])));
}

// ---------------- split qkv -> per-head bf16 Q,K and modulated V^T ----------
__global__ void k_qkvprep(const float* __restrict__ qkv, const float* __restrict__ sel,
                          bf16_t* __restrict__ Qb, bf16_t* __restrict__ Kb,
                          bf16_t* __restrict__ Vt) {
  const int idx = blockIdx.x * 256 + threadIdx.x;
  if (idx >= MROWS * HIDN) return;
  const int m = idx / HIDN, c = idx % HIDN;
  const int h = c / HDIM, d = c % HDIM;
  const int b = m / TSEQ, t = m % TSEQ;
  const size_t base = (size_t)m * (3 * HIDN);
  const float q = qkv[base + c];
  const float k = qkv[base + HIDN + c];
  const float v = qkv[base + 2 * HIDN + c] * (0.3f + 0.7f * sel[m]);
  const size_t bh = (size_t)(b * NHEAD + h);
  Qb[(bh * TSEQ + t) * HDIM + d] = f2bf(q);
  Kb[(bh * TSEQ + t) * HDIM + d] = f2bf(k);
  Vt[(bh * HDIM + d) * TSEQ + t] = f2bf(v);   // transposed for WMMA B fragment
}

// ---------------- causal attention: 1 wave per (b,h,16-row q tile) ----------
__global__ void k_attn(const bf16_t* __restrict__ Qb, const bf16_t* __restrict__ Kb,
                       const bf16_t* __restrict__ Vt, bf16_t* __restrict__ ctx) {
  extern __shared__ char smem[];
  float*  sS = (float*)smem;                        // [16][TSEQ] f32 scores
  bf16_t* sP = (bf16_t*)(smem + 16 * TSEQ * 4);     // [16][TSEQ] bf16 probs
  const int lane = threadIdx.x;
  const int i = blockIdx.x;       // q tile
  const int h = blockIdx.y;
  const int b = blockIdx.z;
  const int half = lane >> 4, lh = lane & 15;
  const bf16_t* Q  = Qb + ((size_t)(b * NHEAD + h) * TSEQ) * HDIM;
  const bf16_t* Kp = Kb + ((size_t)(b * NHEAD + h) * TSEQ) * HDIM;
  const bf16_t* V  = Vt + ((size_t)(b * NHEAD + h) * HDIM) * TSEQ;
  // zero P (covers causal mask + K-chunk padding)
  unsigned* sPu = (unsigned*)sP;
  for (int idx = lane; idx < 16 * TSEQ / 2; idx += 32) sPu[idx] = 0u;
  const int qg0 = i * 16;
  // S = Q K^T * 1/8 for all causal k-tiles
  for (int j = 0; j <= i; ++j) {
    v8f acc = {};
#pragma unroll
    for (int kk = 0; kk < HDIM; kk += 32) {
      v8bf alo = *(const v8bf*)(Q + (size_t)(qg0 + lh) * HDIM + kk + half * 16);
      v8bf ahi = *(const v8bf*)(Q + (size_t)(qg0 + lh) * HDIM + kk + half * 16 + 8);
      v16bf bf = *(const v16bf*)(Kp + (size_t)(j * 16 + lh) * HDIM + kk + half * 16);
      acc = wmma_bf16(pack16(alo, ahi), bf, acc);
    }
#pragma unroll
    for (int r = 0; r < 8; ++r)
      sS[(half * 8 + r) * TSEQ + j * 16 + lh] = acc[r] * 0.125f;
  }
  __syncthreads();
  // exact causal softmax, one row per lane (rows 0..15)
  if (lane < 16) {
    const int r = lane, qg = qg0 + r, len = qg + 1;
    float mxv = -3.4e38f;
    for (int t = 0; t < len; ++t) mxv = fmaxf(mxv, sS[r * TSEQ + t]);
    float sum = 0.f;
    for (int t = 0; t < len; ++t) { float e = __expf(sS[r * TSEQ + t] - mxv); sS[r * TSEQ + t] = e; sum += e; }
    const float inv = 1.f / sum;
    for (int t = 0; t < len; ++t) sP[r * TSEQ + t] = f2bf(sS[r * TSEQ + t] * inv);
  }
  __syncthreads();
  // O = P @ V   (P from LDS, V pre-transposed [d][t])
  const int chunks = (i + 2) >> 1;   // ceil(((i+1)*16)/32)
  for (int nt = 0; nt < 4; ++nt) {
    v8f acc = {};
    for (int c = 0; c < chunks; ++c) {
      v8bf alo = *(const v8bf*)(sP + (size_t)lh * TSEQ + c * 32 + half * 16);
      v8bf ahi = *(const v8bf*)(sP + (size_t)lh * TSEQ + c * 32 + half * 16 + 8);
      v16bf bf = *(const v16bf*)(V + (size_t)(nt * 16 + lh) * TSEQ + c * 32 + half * 16);
      acc = wmma_bf16(pack16(alo, ahi), bf, acc);
    }
#pragma unroll
    for (int r = 0; r < 8; ++r) {
      const int tq = qg0 + half * 8 + r;
      ctx[((size_t)(b * TSEQ + tq)) * HIDN + h * HDIM + nt * 16 + lh] = f2bf(acc[r]);
    }
  }
}

// ======================= host orchestration =================================
extern "C" void kernel_launch(void* const* d_in, const int* in_sizes, int n_in,
                              void* d_out, int out_size, void* d_ws, size_t ws_size,
                              hipStream_t stream) {
  (void)in_sizes; (void)n_in; (void)out_size; (void)ws_size;
  const float* x      = (const float*)d_in[0];
  const float* ln1_g  = (const float*)d_in[1];
  const float* ln1_b  = (const float*)d_in[2];
  const float* attn_w = (const float*)d_in[3];
  const float* attn_b = (const float*)d_in[4];
  const float* proj_w = (const float*)d_in[5];
  const float* proj_b = (const float*)d_in[6];
  const float* mln_g  = (const float*)d_in[7];
  const float* mln_b  = (const float*)d_in[8];
  const float* rms_w  = (const float*)d_in[9];
  const float* in_w   = (const float*)d_in[10];
  const float* conv_w = (const float*)d_in[11];
  const float* conv_b = (const float*)d_in[12];
  const float* xproj_w= (const float*)d_in[13];
  const float* dt_w   = (const float*)d_in[14];
  const float* dt_b   = (const float*)d_in[15];
  const float* A_log  = (const float*)d_in[16];
  const float* Dp     = (const float*)d_in[17];
  const float* out_w  = (const float*)d_in[18];
  const float* sel_w  = (const float*)d_in[19];
  const float* sel_b  = (const float*)d_in[20];
  const float* ln2_g  = (const float*)d_in[21];
  const float* ln2_b  = (const float*)d_in[22];
  const float* fc1_w  = (const float*)d_in[23];
  const float* fc1_b  = (const float*)d_in[24];
  const float* fc2_w  = (const float*)d_in[25];
  const float* fc2_b  = (const float*)d_in[26];
  float* out = (float*)d_out;

  char* p = (char*)d_ws;
  auto alloc = [&](size_t bytes) -> void* {
    void* r = (void*)p;
    p += (bytes + 255) & ~(size_t)255;
    return r;
  };
  // bf16 transposed weights [N, Kpad]
  bf16_t* wt_attn  = (bf16_t*)alloc((size_t)2304 * 768  * 2);
  bf16_t* wt_in    = (bf16_t*)alloc((size_t)3072 * 768  * 2);
  bf16_t* wt_xproj = (bf16_t*)alloc((size_t)80   * 1536 * 2);
  bf16_t* wt_dt    = (bf16_t*)alloc((size_t)1536 * 64   * 2);
  bf16_t* wt_out   = (bf16_t*)alloc((size_t)768  * 1536 * 2);
  bf16_t* wt_proj  = (bf16_t*)alloc((size_t)768  * 768  * 2);
  bf16_t* wt_fc1   = (bf16_t*)alloc((size_t)3072 * 768  * 2);
  bf16_t* wt_fc2   = (bf16_t*)alloc((size_t)768  * 3072 * 2);
  // activations
  bf16_t* xn_bf  = (bf16_t*)alloc((size_t)MROWS * HIDN * 2);
  float*  mx     = (float*) alloc((size_t)MROWS * HIDN * 4);
  bf16_t* h_bf   = (bf16_t*)alloc((size_t)MROWS * HIDN * 2);
  float*  qkv    = (float*) alloc((size_t)MROWS * 3 * HIDN * 4);
  float*  projf  = (float*) alloc((size_t)MROWS * 2 * INTC * 4);
  float*  hsconv = (float*) alloc((size_t)MROWS * INTC * 4);
  bf16_t* hsc_bf = (bf16_t*)alloc((size_t)MROWS * INTC * 2);
  float*  ssm    = (float*) alloc((size_t)MROWS * XDC * 4);
  bf16_t* dt_bf  = (bf16_t*)alloc((size_t)MROWS * 64 * 2);
  float*  dtfull = (float*) alloc((size_t)MROWS * INTC * 4);
  bf16_t* y_bf   = (bf16_t*)alloc((size_t)MROWS * INTC * 2);
  float*  mo     = (float*) alloc((size_t)MROWS * HIDN * 4);
  float*  sel    = (float*) alloc((size_t)MROWS * 4);
  bf16_t* Qb     = (bf16_t*)alloc((size_t)BBATCH * NHEAD * TSEQ * HDIM * 2);
  bf16_t* Kb     = (bf16_t*)alloc((size_t)BBATCH * NHEAD * TSEQ * HDIM * 2);
  bf16_t* Vt     = (bf16_t*)alloc((size_t)BBATCH * NHEAD * HDIM * TSEQ * 2);
  bf16_t* ctx_bf = (bf16_t*)alloc((size_t)MROWS * HIDN * 2);
  float*  x2     = (float*) alloc((size_t)MROWS * HIDN * 4);
  bf16_t* h2_bf  = (bf16_t*)alloc((size_t)MROWS * HIDN * 2);
  bf16_t* fc1_bf = (bf16_t*)alloc((size_t)MROWS * 3072 * 2);

  auto wcast = [&](const float* W, bf16_t* Wt, int K, int N, int Kpad) {
    long long tot = (long long)N * Kpad;
    int blocks = (int)((tot + 255) / 256);
    k_wcast<<<blocks, 256, 0, stream>>>(W, Wt, K, N, Kpad);
  };
  wcast(attn_w,  wt_attn,  768,  2304, 768);
  wcast(in_w,    wt_in,    768,  3072, 768);
  wcast(xproj_w, wt_xproj, 1536, 80,   1536);
  wcast(dt_w,    wt_dt,    48,   1536, 64);     // pad K 48 -> 64
  wcast(out_w,   wt_out,   1536, 768,  1536);
  wcast(proj_w,  wt_proj,  768,  768,  768);
  wcast(fc1_w,   wt_fc1,   768,  3072, 768);
  wcast(fc2_w,   wt_fc2,   3072, 768,  3072);

  const dim3 gblock(32 * GEMM_WAVES);
  auto ggrid = [](int N) { return dim3(N / 16, MROWS / (16 * GEMM_WAVES)); };
  auto bpanel = [](int K) -> size_t { return (size_t)16 * K * 2; }; // LDS panel bytes

  // LN1 + mamba LN + rms-norm
  k_ln1<<<MROWS, 256, 0, stream>>>(x, ln1_g, ln1_b, mln_g, mln_b, rms_w,
                                   xn_bf, mx, h_bf);
  // qkv = ln1(x) @ attn_w + attn_b
  k_gemm<0,1,0,0><<<ggrid(3 * HIDN), gblock, bpanel(768), stream>>>(
      xn_bf, wt_attn, (void*)qkv, attn_b, nullptr, 3 * HIDN, 768);
  // mamba in-proj
  k_gemm<0,0,0,0><<<ggrid(2 * INTC), gblock, bpanel(768), stream>>>(
      h_bf, wt_in, (void*)projf, nullptr, nullptr, 2 * INTC, 768);
  // causal depthwise conv + silu
  k_conv<<<dim3(INTC / 256, MROWS), 256, 0, stream>>>(projf, conv_w, conv_b,
                                                      hsconv, hsc_bf);
  // ssm = hs @ xproj_w
  k_gemm<0,0,0,0><<<ggrid(XDC), gblock, bpanel(1536), stream>>>(
      hsc_bf, wt_xproj, (void*)ssm, nullptr, nullptr, XDC, 1536);
  // dt slice, padded K
  k_dtprep<<<(MROWS * 64) / 256, 256, 0, stream>>>(ssm, dt_bf);
  // dt = softplus(dt @ dt_w + dt_b)
  k_gemm<1,1,0,0><<<ggrid(INTC), gblock, bpanel(64), stream>>>(
      dt_bf, wt_dt, (void*)dtfull, dt_b, nullptr, INTC, 64);
  // selective scan + D skip + silu(gate)
  k_scan<<<(BBATCH * INTC) / 256, 256, 0, stream>>>(dtfull, ssm, hsconv, projf,
                                                    A_log, Dp, y_bf);
  // mo = mln(x) + y @ out_w
  k_gemm<0,0,1,0><<<ggrid(HIDN), gblock, bpanel(1536), stream>>>(
      y_bf, wt_out, (void*)mo, nullptr, mx, HIDN, 1536);
  // sel = sigmoid(clip(mo) @ sel_w + sel_b)
  k_sel<<<MROWS, 256, 0, stream>>>(mo, sel_w, sel_b, sel);
  // per-head Q, K, modulated V^T
  k_qkvprep<<<(MROWS * HIDN) / 256, 256, 0, stream>>>(qkv, sel, Qb, Kb, Vt);
  // causal attention (WMMA S and P*V, LDS softmax)
  const size_t attn_lds = (size_t)16 * TSEQ * 4 + (size_t)16 * TSEQ * 2; // 96 KB
  k_attn<<<dim3(TSEQ / 16, NHEAD, BBATCH), 32, attn_lds, stream>>>(Qb, Kb, Vt, ctx_bf);
  // x2 = x + ctx @ proj_w + proj_b
  k_gemm<0,1,1,0><<<ggrid(HIDN), gblock, bpanel(768), stream>>>(
      ctx_bf, wt_proj, (void*)x2, proj_b, x, HIDN, 768);
  // MLP
  k_ln2<<<MROWS, 256, 0, stream>>>(x2, ln2_g, ln2_b, h2_bf);
  k_gemm<2,1,0,1><<<ggrid(4 * HIDN), gblock, bpanel(768), stream>>>(
      h2_bf, wt_fc1, (void*)fc1_bf, fc1_b, nullptr, 4 * HIDN, 768);
  k_gemm<0,1,1,0><<<ggrid(HIDN), gblock, bpanel(3072), stream>>>(
      fc1_bf, wt_fc2, (void*)out, fc2_b, x2, HIDN, 3072);
}
